// GroupQueryAttention_61572651155728
// MI455X (gfx1250) — compile-verified, matrix operand order checked by reading, not run on hardware
//
#include <hip/hip_runtime.h>

typedef __attribute__((ext_vector_type(16))) _Float16 v16h;
typedef __attribute__((ext_vector_type(8)))  _Float16 h8;
typedef __attribute__((ext_vector_type(8)))  float    v8f;

static constexpr int S   = 4096;
static constexpr int Dm  = 1024;
static constexpr int Hn  = 16;
static constexpr int KVH = 4;
static constexpr int HD  = 64;
static constexpr int NKV = KVH * HD;   // 256

// ---------------------------------------------------------------------------
// Fragment loader for the CDNA5 16-bit A/B WMMA layout (ISA 7.12.2).
// Lane l (l16 = l&15, hi = l>>4) holds row/col l16 with K elements
// [hi*8 .. hi*8+7] and [16+hi*8 .. 16+hi*8+7].
// `rowptr` points at this lane's 32-wide contiguous K window (16B aligned).
// ---------------------------------------------------------------------------
__device__ inline v16h load_frag(const _Float16* rowptr, int hi) {
  h8 x = *(const h8*)(rowptr + hi * 8);
  h8 y = *(const h8*)(rowptr + 16 + hi * 8);
  v16h r;
#pragma unroll
  for (int j = 0; j < 8; ++j) { r[j] = x[j]; r[j + 8] = y[j]; }
  return r;
}

// ---------------------------------------------------------------------------
// Elementwise f32 -> f16 convert
// ---------------------------------------------------------------------------
__global__ void convert_f16(const float* __restrict__ in, _Float16* __restrict__ out, int n) {
  int i = blockIdx.x * blockDim.x + threadIdx.x;
  if (i < n) out[i] = (_Float16)in[i];
}

// W [Kd x Nd] f32 (row-major)  ->  Wt [Nd x Kd] f16 (row-major)
__global__ void transpose_f16(const float* __restrict__ W, _Float16* __restrict__ Wt,
                              int Kd, int Nd) {
  int i = blockIdx.x * blockDim.x + threadIdx.x;
  if (i >= Kd * Nd) return;
  int k = i / Nd, n = i - k * Nd;
  Wt[(size_t)n * Kd + k] = (_Float16)W[i];
}

// ---------------------------------------------------------------------------
// WMMA GEMM:  Y[M x N] = A[M x Kd] (f16, row-major) @ B + bias
//   B supplied transposed: Bt[N x Kd] f16 row-major.
//   MODE 0: store f16 row-major (ld = N)
//   MODE 1: store f16 transposed (Y^T, ld = M)     -- used for V
//   MODE 2: store f32 row-major (ld = N)           -- final output
// One wave computes a 32(M) x 64(N) tile; 8 waves / 256-thread block.
// grid = (N/64, M/256)
// ---------------------------------------------------------------------------
template <int MODE>
__global__ __launch_bounds__(256) void gemm_wmma(const _Float16* __restrict__ A,
                                                 const _Float16* __restrict__ Bt,
                                                 const float* __restrict__ bias,
                                                 void* __restrict__ Out,
                                                 int M, int N, int Kd) {
  const int lane = threadIdx.x & 31;
  const int wave = threadIdx.x >> 5;
  const int l16  = lane & 15;
  const int hi   = lane >> 4;
  const int m0   = (blockIdx.y * 8 + wave) * 32;
  const int n0   = blockIdx.x * 64;

  const _Float16* arow0 = A  + (size_t)(m0 + l16) * Kd;
  const _Float16* arow1 = A  + (size_t)(m0 + 16 + l16) * Kd;
  const _Float16* brow  = Bt + (size_t)(n0 + l16) * Kd;

  v8f acc0[4], acc1[4];
#pragma unroll
  for (int nt = 0; nt < 4; ++nt) {
    v8f z = {};
    acc0[nt] = z;
    acc1[nt] = z;
  }

  for (int k0 = 0; k0 < Kd; k0 += 32) {
    v16h a0 = load_frag(arow0 + k0, hi);
    v16h a1 = load_frag(arow1 + k0, hi);
    if (k0 + 32 < Kd) {
      __builtin_prefetch(arow0 + k0 + 64, 0, 1);  // global_prefetch_b8
      __builtin_prefetch(arow1 + k0 + 64, 0, 1);
    }
#pragma unroll
    for (int nt = 0; nt < 4; ++nt) {
      v16h b = load_frag(brow + (size_t)(nt * 16) * Kd + k0, hi);
      acc0[nt] = __builtin_amdgcn_wmma_f32_16x16x32_f16(false, a0, false, b,
                                                        (short)0, acc0[nt], false, false);
      acc1[nt] = __builtin_amdgcn_wmma_f32_16x16x32_f16(false, a1, false, b,
                                                        (short)0, acc1[nt], false, false);
    }
  }

  // C/D layout: element (m = i + 8*hi, n = l16)
#pragma unroll
  for (int nt = 0; nt < 4; ++nt) {
    int n = n0 + nt * 16 + l16;
    float bv = bias[n];
#pragma unroll
    for (int i = 0; i < 8; ++i) {
      int ma = m0 + i + 8 * hi;
      int mb = ma + 16;
      float va = acc0[nt][i] + bv;
      float vb = acc1[nt][i] + bv;
      if (MODE == 0) {
        ((_Float16*)Out)[(size_t)ma * N + n] = (_Float16)va;
        ((_Float16*)Out)[(size_t)mb * N + n] = (_Float16)vb;
      } else if (MODE == 1) {
        ((_Float16*)Out)[(size_t)n * M + ma] = (_Float16)va;
        ((_Float16*)Out)[(size_t)n * M + mb] = (_Float16)vb;
      } else {
        ((float*)Out)[(size_t)ma * N + n] = va;
        ((float*)Out)[(size_t)mb * N + n] = vb;
      }
    }
  }
}

// ---------------------------------------------------------------------------
// Flash attention, transposed formulation, TWO query tiles per wave.
//   Q  : [S][1024] f16,  Km : [S][256] f16,  Vt : [256][S] f16,  O : [S][1024] f16
//
// Scores computed as S^T = K_block(A) x Q^T(B): C-layout maps lane -> query,
// per-lane elements -> keys, so online softmax is per-lane (one shfl_xor(16)
// per reduction), and the exp'd score fragments are already in B-fragment
// layout for O^T += V^T(A) x P^T(B) -- no LDS, just an in-lane f32->f16 pack.
// The K-block and V^T A-fragments are shared across both query tiles:
// 16 b128 loads feed 16 WMMAs per 32-key block.
// ---------------------------------------------------------------------------
__global__ __launch_bounds__(256) void attn_kernel(const _Float16* __restrict__ Q,
                                                   const _Float16* __restrict__ Km,
                                                   const _Float16* __restrict__ Vt,
                                                   _Float16* __restrict__ O) {
  const int lane = threadIdx.x & 31;
  const int wave = threadIdx.x >> 5;
  const int l16  = lane & 15;
  const int hi   = lane >> 4;

  const int idx = blockIdx.x * 8 + wave;   // 0..2047
  const int h   = idx >> 7;                // head 0..15
  const int qp2 = idx & 127;               // 32-query pair tile 0..127
  const int kvh = h >> 2;                  // GROUP = 4 (repeat_interleave)

  // Q^T B-fragments for two 16-query tiles (hd halves 0..31 / 32..63)
  v16h qf[2][2];
#pragma unroll
  for (int s = 0; s < 2; ++s) {
    const _Float16* qp = Q + (size_t)(qp2 * 32 + s * 16 + l16) * Dm + h * HD;
    qf[s][0] = load_frag(qp, hi);
    qf[s][1] = load_frag(qp + 32, hi);
  }

  float mrun[2] = {-1e30f, -1e30f};
  float lrun[2] = {0.0f, 0.0f};
  v8f oacc[2][4];  // O^T tiles: (m = hd, n = query)
#pragma unroll
  for (int s = 0; s < 2; ++s)
#pragma unroll
    for (int nt = 0; nt < 4; ++nt) {
      v8f z = {};
      oacc[s][nt] = z;
    }

  const float scale = 0.125f;  // 1/sqrt(64)

  for (int kb = 0; kb < S; kb += 32) {
    // ---- S^T tiles: c[s][t] holds keys kb+t*16+(i+8*hi), query tile s ----
    v8f c[2][2];
#pragma unroll
    for (int t = 0; t < 2; ++t) {
      const _Float16* kp = Km + (size_t)(kb + t * 16 + l16) * NKV + kvh * HD;
      v16h aK0 = load_frag(kp, hi);
      v16h aK1 = load_frag(kp + 32, hi);
#pragma unroll
      for (int s = 0; s < 2; ++s) {
        v8f z = {};
        z       = __builtin_amdgcn_wmma_f32_16x16x32_f16(false, aK0, false, qf[s][0],
                                                         (short)0, z, false, false);
        c[s][t] = __builtin_amdgcn_wmma_f32_16x16x32_f16(false, aK1, false, qf[s][1],
                                                         (short)0, z, false, false);
      }
    }

    // ---- online softmax per query tile (per-lane + one half-swap) ----
    v16h pb[2];
#pragma unroll
    for (int s = 0; s < 2; ++s) {
      float lmax = -1e30f;
#pragma unroll
      for (int i = 0; i < 8; ++i) {
        c[s][0][i] *= scale;
        c[s][1][i] *= scale;
        lmax = fmaxf(lmax, fmaxf(c[s][0][i], c[s][1][i]));
      }
      lmax = fmaxf(lmax, __shfl_xor(lmax, 16, 32));  // combine key halves
      float mnew = fmaxf(mrun[s], lmax);
      float corr = __expf(mrun[s] - mnew);
      mrun[s] = mnew;

      float p0[8], p1[8];
      float ls = 0.0f;
#pragma unroll
      for (int i = 0; i < 8; ++i) {
        p0[i] = __expf(c[s][0][i] - mnew);
        p1[i] = __expf(c[s][1][i] - mnew);
        ls += p0[i] + p1[i];
      }
      ls += __shfl_xor(ls, 16, 32);
      lrun[s] = lrun[s] * corr + ls;

#pragma unroll
      for (int nt = 0; nt < 4; ++nt)
#pragma unroll
        for (int i = 0; i < 8; ++i) oacc[s][nt][i] *= corr;

      // pack P^T B-fragment: c0[j] -> K=hi*8+j, c1[j] -> K=16+hi*8+j
#pragma unroll
      for (int j = 0; j < 8; ++j) {
        pb[s][j]     = (_Float16)p0[j];
        pb[s][j + 8] = (_Float16)p1[j];
      }
    }

    // ---- O^T += V^T(A) x P^T(B); V^T fragments shared by both tiles ----
#pragma unroll
    for (int nt = 0; nt < 4; ++nt) {
      const _Float16* vp = Vt + (size_t)(kvh * HD + nt * 16 + l16) * S + kb;
      v16h av = load_frag(vp, hi);
#pragma unroll
      for (int s = 0; s < 2; ++s)
        oacc[s][nt] = __builtin_amdgcn_wmma_f32_16x16x32_f16(false, av, false, pb[s],
                                                             (short)0, oacc[s][nt], false, false);
    }
  }

  // ---- normalize and store: O^T element (m = hd, n = query) ----
#pragma unroll
  for (int s = 0; s < 2; ++s) {
    float rinv = 1.0f / lrun[s];
    const int srow = qp2 * 32 + s * 16 + l16;
    _Float16* op = O + (size_t)srow * Dm + h * HD + 8 * hi;
#pragma unroll
    for (int nt = 0; nt < 4; ++nt) {
      h8 v;
#pragma unroll
      for (int i = 0; i < 8; ++i) v[i] = (_Float16)(oacc[s][nt][i] * rinv);
      *(h8*)(op + nt * 16) = v;
    }
  }
}

// ---------------------------------------------------------------------------
// Host launch
// ---------------------------------------------------------------------------
extern "C" void kernel_launch(void* const* d_in, const int* in_sizes, int n_in,
                              void* d_out, int out_size, void* d_ws, size_t ws_size,
                              hipStream_t stream) {
  (void)in_sizes; (void)n_in; (void)out_size; (void)ws_size;
  const float* X   = (const float*)d_in[0];
  const float* q_w = (const float*)d_in[1];
  const float* q_b = (const float*)d_in[2];
  const float* k_w = (const float*)d_in[3];
  const float* k_b = (const float*)d_in[4];
  const float* v_w = (const float*)d_in[5];
  const float* v_b = (const float*)d_in[6];
  const float* o_w = (const float*)d_in[7];
  const float* o_b = (const float*)d_in[8];
  float* out = (float*)d_out;

  // Carve workspace (f16 buffers), 256B aligned. Total ~33 MB.
  char* ws = (char*)d_ws;
  auto carve = [&](size_t bytes) {
    char* p = ws;
    ws += (bytes + 255) & ~(size_t)255;
    return p;
  };
  _Float16* Xh  = (_Float16*)carve((size_t)S * Dm * 2);    // X in f16
  _Float16* qwt = (_Float16*)carve((size_t)Dm * Dm * 2);   // q_w^T
  _Float16* kwt = (_Float16*)carve((size_t)Dm * NKV * 2);  // k_w^T
  _Float16* vwt = (_Float16*)carve((size_t)Dm * NKV * 2);  // v_w^T
  _Float16* owt = (_Float16*)carve((size_t)Dm * Dm * 2);   // o_w^T
  _Float16* Qh  = (_Float16*)carve((size_t)S * Dm * 2);    // Q [S][1024]
  _Float16* Kh  = (_Float16*)carve((size_t)S * NKV * 2);   // K [S][256]
  _Float16* Vth = (_Float16*)carve((size_t)NKV * S * 2);   // V^T [256][S]
  _Float16* Ah  = (_Float16*)carve((size_t)S * Dm * 2);    // attention out [S][1024]

  // 1) Convert / transpose inputs to f16
  {
    int n = S * Dm;
    convert_f16<<<(n + 255) / 256, 256, 0, stream>>>(X, Xh, n);
  }
  transpose_f16<<<(Dm * Dm  + 255) / 256, 256, 0, stream>>>(q_w, qwt, Dm, Dm);
  transpose_f16<<<(Dm * NKV + 255) / 256, 256, 0, stream>>>(k_w, kwt, Dm, NKV);
  transpose_f16<<<(Dm * NKV + 255) / 256, 256, 0, stream>>>(v_w, vwt, Dm, NKV);
  transpose_f16<<<(Dm * Dm  + 255) / 256, 256, 0, stream>>>(o_w, owt, Dm, Dm);

  // 2) Projections (WMMA GEMMs), 32x64 tile per wave -> grid.y = M/256
  gemm_wmma<0><<<dim3(Dm / 64,  S / 256), 256, 0, stream>>>(Xh, qwt, q_b, Qh,  S, Dm,  Dm);
  gemm_wmma<0><<<dim3(NKV / 64, S / 256), 256, 0, stream>>>(Xh, kwt, k_b, Kh,  S, NKV, Dm);
  gemm_wmma<1><<<dim3(NKV / 64, S / 256), 256, 0, stream>>>(Xh, vwt, v_b, Vth, S, NKV, Dm);

  // 3) Flash attention: 16 heads x 128 query-pair tiles = 2048 waves, 8/block
  attn_kernel<<<(Hn * (S / 32)) / 8, 256, 0, stream>>>(Qh, Kh, Vth, Ah);

  // 4) Output projection -> f32
  gemm_wmma<2><<<dim3(Dm / 64, S / 256), 256, 0, stream>>>(Ah, owt, o_b, out, S, Dm, Dm);
}
